// CFQModel_2731599200909
// MI455X (gfx1250) — matrix operand order repaired and checked by reading, not compiled, not run on hardware
//
#include <hip/hip_runtime.h>
#include <hip/hip_bf16.h>

// ---------------- problem constants ----------------
#define Bsz 32
#define Ssz 64
#define Tsz 100
#define Esz 128
#define Hsz 256
#define Gsz 1024          // 4*H
#define VSsz 32000
#define SOS_TOK 1

// ---------------- WMMA types ----------------
typedef __attribute__((ext_vector_type(16))) __bf16 v16bf;
typedef __attribute__((ext_vector_type(8)))  float  f32x8;
typedef __attribute__((ext_vector_type(4)))  int    v4i;

union ABFrag {
    v16bf bf;
    uint4 q[2];
};

__device__ __forceinline__ unsigned short f2bf(float f) {
    unsigned int u = __float_as_uint(f);
    u = (u + 0x7FFFu + ((u >> 16) & 1u)) >> 16;   // round-to-nearest-even
    return (unsigned short)u;
}
__device__ __forceinline__ float bf2f(unsigned short u) {
    return __uint_as_float(((unsigned int)u) << 16);
}
__device__ __forceinline__ float sigmoidf(float x) {
    return 1.0f / (1.0f + __expf(-x));
}

// A fragment: 16x32 bf16 tile, rows m0..m0+15, K = kstart..kstart+31.
__device__ __forceinline__ v16bf load_a_frag(const unsigned short* base, int stride,
                                             int m0, int kstart, int lane) {
    int r  = m0 + (lane & 15);
    int k0 = kstart + ((lane >> 4) << 4);
    ABFrag f;
    f.q[0] = *(const uint4*)(base + r * stride + k0);
    f.q[1] = *(const uint4*)(base + r * stride + k0 + 8);
    return f.bf;
}

// B fragment: 32x16 bf16 tile of W^T; per-lane 32 contiguous bytes of row n of W.
__device__ __forceinline__ v16bf load_b_frag(const unsigned short* base, int stride,
                                             int n0, int kstart, int lane) {
    int n  = n0 + (lane & 15);
    int k0 = kstart + ((lane >> 4) << 4);
    ABFrag f;
    const uint4* p = (const uint4*)(base + n * stride + k0);
    f.q[0] = p[0];
    f.q[1] = p[1];
    return f.bf;
}

#define WMMA_BF16(a, b, c) \
    __builtin_amdgcn_wmma_f32_16x16x32_bf16(false, (a), false, (b), (short)0, (c), false, false)

// ---------------- async global->LDS (CDNA5 path, guarded) ----------------
#if __has_builtin(__builtin_amdgcn_global_load_async_to_lds_b128) && \
    __has_builtin(__builtin_amdgcn_s_wait_asynccnt)
#define USE_ASYNC_LDS 1
#else
#define USE_ASYNC_LDS 0
#endif

__device__ __forceinline__ void copy16_g2l(unsigned short* ldst, const unsigned short* gsrc) {
#if USE_ASYNC_LDS
    __builtin_amdgcn_global_load_async_to_lds_b128((v4i*)gsrc, (v4i*)ldst, 0, 0);
#else
    *(uint4*)ldst = *(const uint4*)gsrc;
#endif
}
__device__ __forceinline__ void async_lds_wait() {
#if USE_ASYNC_LDS
    __builtin_amdgcn_s_wait_asynccnt(0);
#endif
}

// ---------------- prep kernels ----------------
__global__ void k_convert_bf16(const float* __restrict__ src,
                               unsigned short* __restrict__ dst, int n) {
    for (int i = blockIdx.x * blockDim.x + threadIdx.x; i < n; i += gridDim.x * blockDim.x)
        dst[i] = f2bf(src[i]);
}

__global__ void k_gather_enc(const int* __restrict__ qids,
                             const float* __restrict__ emb,
                             unsigned short* __restrict__ Xenc) {
    const int total = Ssz * Bsz * Esz;
    for (int i = blockIdx.x * blockDim.x + threadIdx.x; i < total; i += gridDim.x * blockDim.x) {
        int e = i & (Esz - 1);
        int b = (i >> 7) & (Bsz - 1);
        int s = i >> 12;
        int tok = qids[b * Ssz + s];
        Xenc[i] = f2bf(emb[tok * Esz + e]);
    }
}

__global__ void k_gather_dec(const int* __restrict__ sids,
                             const float* __restrict__ emb,
                             unsigned short* __restrict__ Edec) {
    const int total = Tsz * Bsz * Esz;
    for (int i = blockIdx.x * blockDim.x + threadIdx.x; i < total; i += gridDim.x * blockDim.x) {
        int e = i & (Esz - 1);
        int b = (i >> 7) & (Bsz - 1);
        int t = i >> 12;
        int tok = (t == 0) ? SOS_TOK : sids[b * Tsz + (t - 1)];
        Edec[i] = f2bf(emb[tok * Esz + e]);
    }
}

// Per-wave gate GEMM fragment: 4 N-tiles (acc[4]) sharing one A fragment per kc.
// mt = wave&1 (M-tile), nt(j) = (wave>>1) + 16*j.
#define GATE_SEGMENT(ASRC, ASTRIDE, AKOFF, WPTR, WSTRIDE, WKOFF, NKC)            \
    _Pragma("unroll 1")                                                          \
    for (int kc = 0; kc < (NKC); ++kc) {                                         \
        v16bf a = load_a_frag((ASRC), (ASTRIDE), mt * 16, (AKOFF) + kc * 32, lane); \
        _Pragma("unroll")                                                        \
        for (int j = 0; j < 4; ++j) {                                            \
            v16bf b = load_b_frag((WPTR), (WSTRIDE), nt0 * 16 + 256 * j,         \
                                  (WKOFF) + kc * 32, lane);                      \
            acc[j] = WMMA_BF16(a, b, acc[j]);                                    \
        }                                                                        \
    }

// ---------------- encoder: persistent 1-WG, 2-layer LSTM over S steps ----------------
__global__ __launch_bounds__(1024) void k_encoder(
    const unsigned short* __restrict__ Xenc,  // [S][B][E] bf16
    const unsigned short* __restrict__ Wih0,  // [G][E]
    const unsigned short* __restrict__ Whh0,  // [G][H]
    const float* __restrict__ b0,
    const unsigned short* __restrict__ Wih1,  // [G][H]
    const unsigned short* __restrict__ Whh1,  // [G][H]
    const float* __restrict__ b1,
    float* __restrict__ EncOut)               // [B][S][H] f32
{
    __shared__ __align__(16) unsigned short s_gates[Bsz * Gsz];  // 64KB
    __shared__ __align__(16) unsigned short s_h0[Bsz * Hsz];     // 16KB
    __shared__ __align__(16) unsigned short s_h1[Bsz * Hsz];     // 16KB

    const int tid  = threadIdx.x;
    const int lane = tid & 31;
    const int wave = tid >> 5;
    const int mt   = wave & 1;
    const int nt0  = wave >> 1;

    for (int i = tid; i < Bsz * Hsz; i += 1024) { s_h0[i] = 0; s_h1[i] = 0; }
    float c0[8], c1[8];
#pragma unroll
    for (int e = 0; e < 8; ++e) { c0[e] = 0.f; c1[e] = 0.f; }
    __syncthreads();

#pragma unroll 1
    for (int s = 0; s < Ssz; ++s) {
        const unsigned short* X = Xenc + s * (Bsz * Esz);

        // ---- layer 0: gates = X @ Wih0^T + h0 @ Whh0^T + b0 ----
        {
            f32x8 acc[4];
#pragma unroll
            for (int j = 0; j < 4; ++j) {
                float bias = b0[nt0 * 16 + 256 * j + (lane & 15)];
#pragma unroll
                for (int r = 0; r < 8; ++r) acc[j][r] = bias;
            }
            GATE_SEGMENT(X, Esz, 0, Wih0, Esz, 0, Esz / 32)
            GATE_SEGMENT(s_h0, Hsz, 0, Whh0, Hsz, 0, Hsz / 32)
#pragma unroll
            for (int j = 0; j < 4; ++j)
#pragma unroll
                for (int r = 0; r < 8; ++r) {
                    int m = mt * 16 + ((lane >> 4) << 3) + r;
                    s_gates[m * Gsz + nt0 * 16 + 256 * j + (lane & 15)] = f2bf(acc[j][r]);
                }
        }
        __syncthreads();

        // elementwise layer 0 (fixed (b,j)->thread mapping keeps c in regs)
#pragma unroll
        for (int e = 0; e < 8; ++e) {
            int u = tid + e * 1024;
            int b = u >> 8, jj = u & 255;
            float ig = bf2f(s_gates[b * Gsz + jj]);
            float fg = bf2f(s_gates[b * Gsz + 256 + jj]);
            float gg = bf2f(s_gates[b * Gsz + 512 + jj]);
            float og = bf2f(s_gates[b * Gsz + 768 + jj]);
            float c2 = sigmoidf(fg) * c0[e] + sigmoidf(ig) * tanhf(gg);
            float h2 = sigmoidf(og) * tanhf(c2);
            c0[e] = c2;
            s_h0[u] = f2bf(h2);
        }
        __syncthreads();

        // ---- layer 1: gates = h0 @ Wih1^T + h1 @ Whh1^T + b1 ----
        {
            f32x8 acc[4];
#pragma unroll
            for (int j = 0; j < 4; ++j) {
                float bias = b1[nt0 * 16 + 256 * j + (lane & 15)];
#pragma unroll
                for (int r = 0; r < 8; ++r) acc[j][r] = bias;
            }
            GATE_SEGMENT(s_h0, Hsz, 0, Wih1, Hsz, 0, Hsz / 32)
            GATE_SEGMENT(s_h1, Hsz, 0, Whh1, Hsz, 0, Hsz / 32)
#pragma unroll
            for (int j = 0; j < 4; ++j)
#pragma unroll
                for (int r = 0; r < 8; ++r) {
                    int m = mt * 16 + ((lane >> 4) << 3) + r;
                    s_gates[m * Gsz + nt0 * 16 + 256 * j + (lane & 15)] = f2bf(acc[j][r]);
                }
        }
        __syncthreads();

        // elementwise layer 1 + write enc_out[b][s][:]
#pragma unroll
        for (int e = 0; e < 8; ++e) {
            int u = tid + e * 1024;
            int b = u >> 8, jj = u & 255;
            float ig = bf2f(s_gates[b * Gsz + jj]);
            float fg = bf2f(s_gates[b * Gsz + 256 + jj]);
            float gg = bf2f(s_gates[b * Gsz + 512 + jj]);
            float og = bf2f(s_gates[b * Gsz + 768 + jj]);
            float c2 = sigmoidf(fg) * c1[e] + sigmoidf(ig) * tanhf(gg);
            float h2 = sigmoidf(og) * tanhf(c2);
            c1[e] = c2;
            s_h1[u] = f2bf(h2);
            EncOut[b * (Ssz * Hsz) + s * Hsz + jj] = h2;
        }
        __syncthreads();
    }
}

// ---------------- decoder recurrence: persistent 1-WG over T steps ----------------
__global__ __launch_bounds__(1024) void k_decoder(
    const float* __restrict__ EncOut,        // [B][S][H]
    const unsigned short* __restrict__ Edec, // [T][B][E] bf16
    const unsigned short* __restrict__ dWih, // [G][E+H]
    const unsigned short* __restrict__ dWhh, // [G][H]
    const float* __restrict__ db,
    unsigned short* __restrict__ Hdec)       // [T][B][H] bf16 out
{
    __shared__ __align__(16) unsigned short s_gates[Bsz * Gsz]; // 64KB
    __shared__ __align__(16) unsigned short s_hbf[Bsz * Hsz];   // 16KB
    __shared__ __align__(16) unsigned short s_ctx[Bsz * Hsz];   // 16KB
    __shared__ __align__(16) float s_attn[Bsz * Ssz];           // 8KB

    const int tid  = threadIdx.x;
    const int lane = tid & 31;
    const int wave = tid >> 5;
    const int mt   = wave & 1;
    const int nt0  = wave >> 1;

    // h init = enc_out[:, S-1, :]  (== h_top of layer-1)
    for (int i = tid; i < Bsz * Hsz; i += 1024) {
        int b = i >> 8, jj = i & 255;
        s_hbf[i] = f2bf(EncOut[b * (Ssz * Hsz) + (Ssz - 1) * Hsz + jj]);
    }
    __syncthreads();

#pragma unroll 1
    for (int t = 0; t < Tsz; ++t) {
        // ---- attention scores ----
        for (int p = tid; p < Bsz * Ssz; p += 1024) {
            int b = p >> 6, ss = p & 63;
            const float4* er = (const float4*)(EncOut + b * (Ssz * Hsz) + ss * Hsz);
            const unsigned short* hr = s_hbf + b * Hsz;
            float acc = 0.f;
#pragma unroll 4
            for (int k = 0; k < Hsz / 4; ++k) {
                float4 a = er[k];
                acc += a.x * bf2f(hr[4 * k + 0]) + a.y * bf2f(hr[4 * k + 1]) +
                       a.z * bf2f(hr[4 * k + 2]) + a.w * bf2f(hr[4 * k + 3]);
            }
            s_attn[p] = acc;
        }
        __syncthreads();

        // ---- softmax over S ----
        if (tid < Bsz) {
            float mx = -1e30f;
            for (int ss = 0; ss < Ssz; ++ss) mx = fmaxf(mx, s_attn[tid * Ssz + ss]);
            float sum = 0.f;
            for (int ss = 0; ss < Ssz; ++ss) {
                float e = __expf(s_attn[tid * Ssz + ss] - mx);
                s_attn[tid * Ssz + ss] = e;
                sum += e;
            }
            float inv = 1.0f / sum;
            for (int ss = 0; ss < Ssz; ++ss) s_attn[tid * Ssz + ss] *= inv;
        }
        __syncthreads();

        // ---- ctx[b][j] = sum_s attn * enc_out ----
        for (int i = tid; i < Bsz * Hsz; i += 1024) {
            int b = i >> 8, jj = i & 255;
            const float* eb = EncOut + b * (Ssz * Hsz) + jj;
            float acc = 0.f;
#pragma unroll 4
            for (int ss = 0; ss < Ssz; ++ss) acc += s_attn[b * Ssz + ss] * eb[ss * Hsz];
            s_ctx[i] = f2bf(acc);
        }
        __syncthreads();

        // ---- gates = [e|ctx] @ dWih^T + h @ dWhh^T + db ----
        {
            const unsigned short* Xe = Edec + t * (Bsz * Esz);
            f32x8 acc[4];
#pragma unroll
            for (int j = 0; j < 4; ++j) {
                float bias = db[nt0 * 16 + 256 * j + (lane & 15)];
#pragma unroll
                for (int r = 0; r < 8; ++r) acc[j][r] = bias;
            }
            GATE_SEGMENT(Xe, Esz, 0, dWih, Esz + Hsz, 0, Esz / 32)       // x[0:128) = emb
            GATE_SEGMENT(s_ctx, Hsz, 0, dWih, Esz + Hsz, Esz, Hsz / 32)  // x[128:384) = ctx
            GATE_SEGMENT(s_hbf, Hsz, 0, dWhh, Hsz, 0, Hsz / 32)          // recurrent
#pragma unroll
            for (int j = 0; j < 4; ++j)
#pragma unroll
                for (int r = 0; r < 8; ++r) {
                    int m = mt * 16 + ((lane >> 4) << 3) + r;
                    s_gates[m * Gsz + nt0 * 16 + 256 * j + (lane & 15)] = f2bf(acc[j][r]);
                }
        }
        __syncthreads();

        // ---- elementwise (reference passes c = 0 each step) ----
#pragma unroll
        for (int e = 0; e < 8; ++e) {
            int u = tid + e * 1024;
            int b = u >> 8, jj = u & 255;
            float ig = bf2f(s_gates[b * Gsz + jj]);
            float gg = bf2f(s_gates[b * Gsz + 512 + jj]);
            float og = bf2f(s_gates[b * Gsz + 768 + jj]);
            float c2 = sigmoidf(ig) * tanhf(gg);
            float h2 = sigmoidf(og) * tanhf(c2);
            unsigned short hb = f2bf(h2);
            s_hbf[u] = hb;
            Hdec[t * (Bsz * Hsz) + u] = hb;
        }
        __syncthreads();
    }
}

// ---------------- logits: [3200,256] x [256,32000] bf16 WMMA GEMM ----------------
// block = 256 threads = 8 waves; block tile = 64 (M) x 128 (N).
// A block (64x256 bf16 = 32KB) is shared by all 8 waves -> staged once into LDS
// (async global->LDS on CDNA5). B streams from L2-resident Wout.
__global__ __launch_bounds__(256) void k_logits(
    const unsigned short* __restrict__ Hdec, // [T*B][H], row m = t*32+b
    const unsigned short* __restrict__ Wout, // [VS][H]
    const float* __restrict__ bout,
    float* __restrict__ out)                 // [B][T][VS]
{
    __shared__ __align__(16) unsigned short s_a[64 * Hsz];   // 32KB

    const int tid  = threadIdx.x;
    const int lane = tid & 31;
    const int wave = tid >> 5;
    const int n0 = (blockIdx.x * 8 + wave) * 16;
    const int m0 = blockIdx.y * 64;

    // stage A rows m0..m0+63 into LDS: 2048 16B chunks over 256 threads
    {
        const unsigned short* Arow = Hdec + (size_t)m0 * Hsz;
#pragma unroll
        for (int c = 0; c < 8; ++c) {
            int chunk = tid + c * 256;
            copy16_g2l(s_a + chunk * 8, Arow + chunk * 8);
        }
        async_lds_wait();
    }
    __syncthreads();

    float bias = bout[n0 + (lane & 15)];
    f32x8 acc[4];
#pragma unroll
    for (int j = 0; j < 4; ++j)
#pragma unroll
        for (int r = 0; r < 8; ++r) acc[j][r] = bias;

#pragma unroll 1
    for (int kc = 0; kc < Hsz / 32; ++kc) {
        v16bf bfrag = load_b_frag(Wout, Hsz, n0, kc * 32, lane);
#pragma unroll
        for (int j = 0; j < 4; ++j) {
            v16bf afrag = load_a_frag(s_a, Hsz, 16 * j, kc * 32, lane);
            acc[j] = WMMA_BF16(afrag, bfrag, acc[j]);
        }
    }

    const int n = n0 + (lane & 15);
#pragma unroll
    for (int j = 0; j < 4; ++j) {
#pragma unroll
        for (int r = 0; r < 8; ++r) {
            int m = m0 + 16 * j + ((lane >> 4) << 3) + r;
            int t = m >> 5;           // m = t*32 + b
            int b = m & 31;
            out[(size_t)b * (Tsz * VSsz) + (size_t)t * VSsz + n] = acc[j][r];
        }
    }
}

// ---------------- launcher ----------------
extern "C" void kernel_launch(void* const* d_in, const int* in_sizes, int n_in,
                              void* d_out, int out_size, void* d_ws, size_t ws_size,
                              hipStream_t stream) {
    (void)in_sizes; (void)n_in; (void)out_size; (void)ws_size;

    const int*   qids      = (const int*)d_in[0];
    const int*   sids      = (const int*)d_in[1];
    const float* enc_embed = (const float*)d_in[2];
    const float* Wih0f     = (const float*)d_in[3];
    const float* Whh0f     = (const float*)d_in[4];
    const float* b0        = (const float*)d_in[5];
    const float* Wih1f     = (const float*)d_in[6];
    const float* Whh1f     = (const float*)d_in[7];
    const float* b1        = (const float*)d_in[8];
    const float* dec_embed = (const float*)d_in[9];
    const float* dWihf     = (const float*)d_in[10];
    const float* dWhhf     = (const float*)d_in[11];
    const float* db        = (const float*)d_in[12];
    const float* Woutf     = (const float*)d_in[13];
    const float* bout      = (const float*)d_in[14];
    float* out = (float*)d_out;

    constexpr size_t nWih0 = (size_t)Gsz * Esz;
    constexpr size_t nWhh0 = (size_t)Gsz * Hsz;
    constexpr size_t nWih1 = (size_t)Gsz * Hsz;
    constexpr size_t nWhh1 = (size_t)Gsz * Hsz;
    constexpr size_t ndWih = (size_t)Gsz * (Esz + Hsz);
    constexpr size_t ndWhh = (size_t)Gsz * Hsz;
    constexpr size_t nWout = (size_t)VSsz * Hsz;
    constexpr size_t nXenc = (size_t)Ssz * Bsz * Esz;
    constexpr size_t nEdec = (size_t)Tsz * Bsz * Esz;
    constexpr size_t nHdec = (size_t)Tsz * Bsz * Hsz;

    unsigned short* p = (unsigned short*)d_ws;
    unsigned short* wWih0 = p; p += nWih0;
    unsigned short* wWhh0 = p; p += nWhh0;
    unsigned short* wWih1 = p; p += nWih1;
    unsigned short* wWhh1 = p; p += nWhh1;
    unsigned short* wdWih = p; p += ndWih;
    unsigned short* wdWhh = p; p += ndWhh;
    unsigned short* wWout = p; p += nWout;
    unsigned short* Xenc  = p; p += nXenc;
    unsigned short* Edec  = p; p += nEdec;
    unsigned short* Hdec  = p; p += nHdec;
    size_t bfBytes = (size_t)((char*)p - (char*)d_ws);
    bfBytes = (bfBytes + 255) & ~(size_t)255;
    float* EncOut = (float*)((char*)d_ws + bfBytes);   // [B][S][H]

    k_convert_bf16<<<512, 256, 0, stream>>>(Wih0f, wWih0, (int)nWih0);
    k_convert_bf16<<<512, 256, 0, stream>>>(Whh0f, wWhh0, (int)nWhh0);
    k_convert_bf16<<<512, 256, 0, stream>>>(Wih1f, wWih1, (int)nWih1);
    k_convert_bf16<<<512, 256, 0, stream>>>(Whh1f, wWhh1, (int)nWhh1);
    k_convert_bf16<<<512, 256, 0, stream>>>(dWihf, wdWih, (int)ndWih);
    k_convert_bf16<<<512, 256, 0, stream>>>(dWhhf, wdWhh, (int)ndWhh);
    k_convert_bf16<<<2048, 256, 0, stream>>>(Woutf, wWout, (int)nWout);

    k_gather_enc<<<512, 256, 0, stream>>>(qids, enc_embed, Xenc);
    k_gather_dec<<<1024, 256, 0, stream>>>(sids, dec_embed, Edec);

    k_encoder<<<1, 1024, 0, stream>>>(Xenc, wWih0, wWhh0, b0, wWih1, wWhh1, b1, EncOut);
    k_decoder<<<1, 1024, 0, stream>>>(EncOut, Edec, wdWih, wdWhh, db, Hdec);

    dim3 grid(VSsz / (8 * 16), (Tsz * Bsz) / 64);   // (250, 50)
    k_logits<<<grid, 256, 0, stream>>>(Hdec, wWout, bout, out);
}
